// MultiHeadSelfAttention_69303592288775
// MI455X (gfx1250) — matrix operand order repaired and checked by reading, not compiled, not run on hardware
//
#include <hip/hip_runtime.h>
#include <hip/hip_bf16.h>

#define BB 4
#define TT 2048
#define CC 1024
#define HH 16
#define HD 64
#define MTOT (BB*TT)   /* 8192 */

typedef __attribute__((ext_vector_type(16))) __bf16 v16bf;
typedef __attribute__((ext_vector_type(8)))  __bf16 v8bf;
typedef __attribute__((ext_vector_type(8)))  float  v8f;
typedef unsigned int u32;
typedef unsigned long long u64;

#define WMMA_BF16(a,b,c) \
  __builtin_amdgcn_wmma_f32_16x16x32_bf16(false,(a),false,(b),(short)0,(c),false,false)

// ---------------- TDM (Tensor Data Mover) support, guarded ----------------
#if defined(__has_builtin)
#if __has_builtin(__builtin_amdgcn_tensor_load_to_lds) && __has_builtin(__builtin_amdgcn_s_wait_tensorcnt)
#define HAVE_TDM 1
#endif
#endif

#ifdef HAVE_TDM
typedef __attribute__((ext_vector_type(4))) u32 u32x4;
typedef __attribute__((ext_vector_type(4))) int i32x4;
typedef __attribute__((ext_vector_type(8))) int i32x8;

// Build a 2D-tile D# (groups 0/1 per CDNA5 ISA) and issue the DMA.
// All sizes in data_size units (dsz_code: 0=1B,1=2B,2=4B,3=8B).
// This toolchain's builtin is the 6-arg (clang-23 / therock-10.0) form:
//   (uint32x4 g0, int32x8 g1, int32x4 g2, int32x4 g3, int32x8 extra, i32 cpol)
static __device__ __forceinline__ void tdm_load_2d(u32 lds_addr, u64 gaddr,
                                                   u32 dsz_code,
                                                   u32 tile_d0, u32 tile_d1,
                                                   u32 tensor_d0, u32 tensor_d1,
                                                   u64 stride0) {
  u32x4 g0;
  g0[0] = 1u;                                              // count=1, user D#
  g0[1] = lds_addr;                                        // lds_addr (bytes)
  g0[2] = (u32)(gaddr & 0xffffffffu);                      // global_addr[31:0]
  g0[3] = (u32)((gaddr >> 32) & 0x01ffffffu) | 0x80000000u;// addr[56:32] | type=2
  i32x8 g1;
  g1[0] = (int)(dsz_code << 16);                           // wg_mask=0, flags=0
  g1[1] = (int)((tensor_d0 & 0xffffu) << 16);              // barrier_addr=0 | td0.lo
  g1[2] = (int)(((tensor_d0 >> 16) & 0xffffu) | ((tensor_d1 & 0xffffu) << 16));
  g1[3] = (int)(((tensor_d1 >> 16) & 0xffffu) | ((tile_d0 & 0xffffu) << 16));
  g1[4] = (int)(tile_d1 & 0xffffu);                        // tile_dim1 | tile_dim2=0
  g1[5] = (int)(u32)(stride0 & 0xffffffffu);               // dim0_stride[31:0]
  g1[6] = (int)((u32)((stride0 >> 32) & 0xffffu));         // dim0_stride[47:32]
  g1[7] = 0;
  i32x4 z4 = {};
  i32x8 z8 = {};
  __builtin_amdgcn_tensor_load_to_lds(g0, g1, z4, z4, z8, 0);
}
static __device__ __forceinline__ u32 lds_off(const void* p) {
  return (u32)(u64)(uintptr_t)p;   // flat LDS address truncates to LDS offset
}
#endif

union BF2 { u32 u; __bf16 h[2]; };

static __device__ __forceinline__ u32 pack_bf2(float a, float b) {
  BF2 t; t.h[0] = (__bf16)a; t.h[1] = (__bf16)b; return t.u;
}

// A-matrix fragment (16x32 bf16): lane<16 holds M=lane, K = {k0..k0+7, k0+16..k0+23}
// lane>=16 holds M=lane-16, K = {k0+8..k0+15, k0+24..k0+31}
static __device__ __forceinline__ v16bf load_a_frag(const __bf16* base, int lda,
                                                    int row, int k0, int lane) {
  const int hi = lane >> 4;
  const __bf16* p = base + row * lda + k0;
  union { v16bf v; v8bf h[2]; } u;
  u.h[0] = *(const v8bf*)(p + hi * 8);
  u.h[1] = *(const v8bf*)(p + 16 + hi * 8);
  return u.v;
}

// B-matrix fragment (32x16 bf16) loaded from a B^T (row = N, col = K) LDS tile:
// lane<16: N=lane, K={k0..k0+7, k0+16..k0+23} ; lane>=16: N=lane-16, K={k0+8..k0+15, k0+24..k0+31}
static __device__ __forceinline__ v16bf load_b_frag(const __bf16* base, int ldb,
                                                    int nrow, int k0, int lane) {
  const int hi = lane >> 4;
  const __bf16* p = base + nrow * ldb + k0;
  union { v16bf v; v8bf h[2]; } u;
  u.h[0] = *(const v8bf*)(p + hi * 8);
  u.h[1] = *(const v8bf*)(p + 16 + hi * 8);
  return u.v;
}

// ---------------- Kernel 1: qkv = x @ Wqkv, scatter to Q (scaled), K, V^T ------------
__global__ __launch_bounds__(128) void qkv_kernel(const float* __restrict__ x,
                                                  const float* __restrict__ Wqkv,
                                                  __bf16* __restrict__ Qb,
                                                  __bf16* __restrict__ Kb,
                                                  __bf16* __restrict__ VT) {
  __shared__ __bf16 As[64 * 32];
  __shared__ __bf16 Bs[64 * 32];
  const int tid  = threadIdx.x;
  const int lane = tid & 31, wv = tid >> 5;
  const int l = lane & 15, hi = lane >> 4;
  const int m0 = blockIdx.x * 64;
  const int n0 = blockIdx.y * 64;

  v8f acc[4] = {};
  const int arow  = tid >> 1;        // 0..63
  const int ahalf = (tid & 1) * 16;  // 0 or 16

  for (int kb = 0; kb < CC; kb += 32) {
    { // stage A (x) tile, fp32 -> bf16
      const float* src = x + (size_t)(m0 + arow) * CC + kb + ahalf;
      __builtin_prefetch(src + 32, 0, 1);
      u32* dst = (u32*)&As[arow * 32 + ahalf];
#pragma unroll
      for (int i = 0; i < 8; i++) dst[i] = pack_bf2(src[2 * i], src[2 * i + 1]);
    }
    { // stage B^T (Wqkv) tile: Bs[n][k] = Wqkv[k * 3C + n]
      const float* src = Wqkv + (size_t)(kb + ahalf) * (3 * CC) + n0 + arow;
      u32* dst = (u32*)&Bs[arow * 32 + ahalf];
#pragma unroll
      for (int i = 0; i < 8; i++)
        dst[i] = pack_bf2(src[(size_t)(2 * i) * (3 * CC)],
                          src[(size_t)(2 * i + 1) * (3 * CC)]);
    }
    __syncthreads();
    v16bf af = load_a_frag(As, 32, wv * 16 + l, 0, lane);
#pragma unroll
    for (int nt = 0; nt < 4; nt++) {
      v16bf bf = load_b_frag(Bs, 32, nt * 16 + l, 0, lane);
      acc[nt] = WMMA_BF16(af, bf, acc[nt]);
    }
    __syncthreads();
  }

  const int which = n0 / CC;        // 0=q, 1=k, 2=v (uniform per block)
  const int cbase = n0 % CC;        // multiple of 64
  const int h     = cbase / HD;     // uniform per block
  const int b     = m0 / TT;        // uniform: 64-row tiles never straddle a batch
  const int t0    = (m0 % TT) + wv * 16 + hi * 8;
  const size_t bh = (size_t)(b * HH + h);
#pragma unroll
  for (int nt = 0; nt < 4; nt++) {
    const int d = nt * 16 + l;
#pragma unroll
    for (int j = 0; j < 8; j++) {
      const int t = t0 + j;
      const float v = acc[nt][j];
      if (which == 0)      Qb[(bh * TT + t) * HD + d] = (__bf16)(v * 0.125f);
      else if (which == 1) Kb[(bh * TT + t) * HD + d] = (__bf16)v;
      else                 VT[(bh * HD + d) * TT + t] = (__bf16)v;
    }
  }
}

// ---------------- Kernel 2: causal flash attention, bf16 WMMA --------------------
__global__ __launch_bounds__(128) void attn_kernel(const __bf16* __restrict__ Qb,
                                                   const __bf16* __restrict__ Kb,
                                                   const __bf16* __restrict__ VT,
                                                   __bf16* __restrict__ AO) {
  __shared__ __bf16 Ks[64 * 64];       // [key][d]   == B^T for S = Q K^T
  __shared__ __bf16 Vs[64 * 64];       // [d][key]   == B^T for O = P V
  __shared__ __bf16 Ps[4][16 * 64];    // per-wave P tile [qrow][key]
  const int tid  = threadIdx.x;
  const int lane = tid & 31, wv = tid >> 5;
  const int l = lane & 15, hi = lane >> 4;
  const int tq0 = blockIdx.x * 64;
  const int bh  = blockIdx.y;
  const __bf16* Q = Qb + (size_t)bh * TT * HD;
  const __bf16* K = Kb + (size_t)bh * TT * HD;
  const __bf16* V = VT + (size_t)bh * HD * TT;

  // Q fragments (16 rows x 64 d) live in registers for the whole loop
  const v16bf qf0 = load_a_frag(Q, HD, tq0 + wv * 16 + l, 0, lane);
  const v16bf qf1 = load_a_frag(Q, HD, tq0 + wv * 16 + l, 32, lane);

  v8f o[4] = {};
  float mrow[8], lrow[8];
#pragma unroll
  for (int j = 0; j < 8; j++) { mrow[j] = -3.0e38f; lrow[j] = 0.f; }

  const int row  = tid >> 1;          // 0..63 (cooperative tile loads)
  const int half = (tid & 1) * 32;    // 0 or 32
  const int ntiles = tq0 / 64 + 1;    // causal: only tiles up to the diagonal

  for (int kt = 0; kt < ntiles; ++kt) {
    const int k0g = kt * 64;
#ifdef HAVE_TDM
    // Tensor Data Mover: K tile is 8KB contiguous (1D), V^T tile is 64 rows of
    // 128B with a 4096B stride (2D). 8-byte elements. TDM ignores EXEC, so a
    // single-lane branch issues it once; wait TENSORcnt before the barrier.
    if (tid == 0) {
      tdm_load_2d(lds_off(&Ks[0]),
                  (u64)(uintptr_t)(K + (size_t)k0g * HD),
                  3u, /*tile*/1024u, 1u, /*tensor*/1024u, 1u, /*stride*/1024ull);
      tdm_load_2d(lds_off(&Vs[0]),
                  (u64)(uintptr_t)(V + k0g),
                  3u, /*tile*/16u, 64u, /*tensor*/512u, 64u, /*stride*/512ull);
      __builtin_amdgcn_s_wait_tensorcnt(0);
    }
#else
    { // cooperative fallback: stage K [64 keys][64 d] and V^T [64 d][64 keys]
      const uint4* ksrc = (const uint4*)(K + (size_t)(k0g + row) * HD + half);
      const uint4* vsrc = (const uint4*)(V + (size_t)row * TT + k0g + half);
      uint4* kdst = (uint4*)&Ks[row * 64 + half];
      uint4* vdst = (uint4*)&Vs[row * 64 + half];
#pragma unroll
      for (int i = 0; i < 4; i++) { kdst[i] = ksrc[i]; vdst[i] = vsrc[i]; }
    }
#endif
    __syncthreads();

    // S = Q K^T (scale already folded into Q), causal mask, track row max
    float mcur[8];
#pragma unroll
    for (int j = 0; j < 8; j++) mcur[j] = -3.0e38f;
    v8f s[4];
#pragma unroll
    for (int nt = 0; nt < 4; ++nt) {
      v8f sc = {};
      v16bf kf0 = load_b_frag(Ks, 64, nt * 16 + l, 0, lane);
      v16bf kf1 = load_b_frag(Ks, 64, nt * 16 + l, 32, lane);
      sc = WMMA_BF16(qf0, kf0, sc);
      sc = WMMA_BF16(qf1, kf1, sc);
      const int kglob = k0g + nt * 16 + l;
#pragma unroll
      for (int j = 0; j < 8; j++) {
        const int qglob = tq0 + wv * 16 + hi * 8 + j;
        const float val = (kglob <= qglob) ? sc[j] : -1.0e30f;
        sc[j] = val;
        mcur[j] = fmaxf(mcur[j], val);
      }
      s[nt] = sc;
    }
    // row max across the 16 lanes holding one row (halves are independent)
#pragma unroll
    for (int j = 0; j < 8; j++) {
      float m = mcur[j];
      m = fmaxf(m, __shfl_xor(m, 1, 16));
      m = fmaxf(m, __shfl_xor(m, 2, 16));
      m = fmaxf(m, __shfl_xor(m, 4, 16));
      m = fmaxf(m, __shfl_xor(m, 8, 16));
      mcur[j] = m;
    }
    float alpha[8], psum[8];
#pragma unroll
    for (int j = 0; j < 8; j++) {
      const float mnew = fmaxf(mrow[j], mcur[j]);
      alpha[j] = __expf(mrow[j] - mnew);
      mrow[j]  = mnew;
      psum[j]  = 0.f;
    }
    // P = exp(S - m), spill to per-wave LDS tile (re-layout C -> A operand)
#pragma unroll
    for (int nt = 0; nt < 4; ++nt) {
#pragma unroll
      for (int j = 0; j < 8; j++) {
        const float p = __expf(s[nt][j] - mrow[j]);
        psum[j] += p;
        Ps[wv][(hi * 8 + j) * 64 + nt * 16 + l] = (__bf16)p;
      }
    }
#pragma unroll
    for (int j = 0; j < 8; j++) {
      float ps = psum[j];
      ps += __shfl_xor(ps, 1, 16);
      ps += __shfl_xor(ps, 2, 16);
      ps += __shfl_xor(ps, 4, 16);
      ps += __shfl_xor(ps, 8, 16);
      lrow[j] = lrow[j] * alpha[j] + ps;
    }
#pragma unroll
    for (int nt = 0; nt < 4; ++nt)
#pragma unroll
      for (int j = 0; j < 8; j++) o[nt][j] *= alpha[j];

    // O += P V  (same-wave LDS RAW, DS ops are in-order per wave)
    v16bf pf0 = load_a_frag(Ps[wv], 64, l, 0, lane);
    v16bf pf1 = load_a_frag(Ps[wv], 64, l, 32, lane);
#pragma unroll
    for (int nt = 0; nt < 4; ++nt) {
      v16bf vf0 = load_b_frag(Vs, 64, nt * 16 + l, 0, lane);
      v16bf vf1 = load_b_frag(Vs, 64, nt * 16 + l, 32, lane);
      o[nt] = WMMA_BF16(pf0, vf0, o[nt]);
      o[nt] = WMMA_BF16(pf1, vf1, o[nt]);
    }
    __syncthreads();
  }

  // normalize and write attention output as bf16 [B*T, C] (head-interleaved)
#pragma unroll
  for (int j = 0; j < 8; j++) lrow[j] = 1.0f / lrow[j];
  const int b = bh / HH, h = bh % HH;
#pragma unroll
  for (int nt = 0; nt < 4; ++nt) {
    const int d = nt * 16 + l;
#pragma unroll
    for (int j = 0; j < 8; j++) {
      const int t = tq0 + wv * 16 + hi * 8 + j;
      AO[((size_t)(b * TT + t)) * CC + h * HD + d] = (__bf16)(o[nt][j] * lrow[j]);
    }
  }
}

// ---------------- Kernel 3: out = AO @ Wout + bout (fp32 out) --------------------
__global__ __launch_bounds__(128) void proj_kernel(const __bf16* __restrict__ AO,
                                                   const float* __restrict__ Wout,
                                                   const float* __restrict__ bout,
                                                   float* __restrict__ out) {
  __shared__ __bf16 As[64 * 32];
  __shared__ __bf16 Bs[64 * 32];
  const int tid  = threadIdx.x;
  const int lane = tid & 31, wv = tid >> 5;
  const int l = lane & 15, hi = lane >> 4;
  const int m0 = blockIdx.x * 64;
  const int n0 = blockIdx.y * 64;

  v8f acc[4] = {};
  const int arow  = tid >> 1;
  const int ahalf = (tid & 1) * 16;

  for (int kb = 0; kb < CC; kb += 32) {
#ifdef HAVE_TDM
    // A tile is already bf16 in global: 64 rows x 64B with 2048B stride -> TDM 2D.
    if (tid == 0) {
      tdm_load_2d(lds_off(&As[0]),
                  (u64)(uintptr_t)(AO + (size_t)m0 * CC + kb),
                  3u, /*tile*/8u, 64u, /*tensor*/256u, 64u, /*stride*/256ull);
    }
#else
    { // A tile is already bf16: plain 16B copies
      const uint4* src = (const uint4*)(AO + (size_t)(m0 + arow) * CC + kb + ahalf);
      uint4* dst = (uint4*)&As[arow * 32 + ahalf];
      dst[0] = src[0]; dst[1] = src[1];
    }
#endif
    { // B^T tile from Wout fp32
      const float* src = Wout + (size_t)(kb + ahalf) * CC + n0 + arow;
      __builtin_prefetch(src + 32 * CC, 0, 1);
      u32* dst = (u32*)&Bs[arow * 32 + ahalf];
#pragma unroll
      for (int i = 0; i < 8; i++)
        dst[i] = pack_bf2(src[(size_t)(2 * i) * CC], src[(size_t)(2 * i + 1) * CC]);
    }
#ifdef HAVE_TDM
    if (tid == 0) __builtin_amdgcn_s_wait_tensorcnt(0);
#endif
    __syncthreads();
    v16bf af = load_a_frag(As, 32, wv * 16 + l, 0, lane);
#pragma unroll
    for (int nt = 0; nt < 4; nt++) {
      v16bf bf = load_b_frag(Bs, 32, nt * 16 + l, 0, lane);
      acc[nt] = WMMA_BF16(af, bf, acc[nt]);
    }
    __syncthreads();
  }
#pragma unroll
  for (int nt = 0; nt < 4; ++nt) {
    const int n = n0 + nt * 16 + l;
    const float bias = bout[n];
#pragma unroll
    for (int j = 0; j < 8; j++) {
      const int m = m0 + wv * 16 + hi * 8 + j;
      out[(size_t)m * CC + n] = acc[nt][j] + bias;
    }
  }
}

extern "C" void kernel_launch(void* const* d_in, const int* in_sizes, int n_in,
                              void* d_out, int out_size, void* d_ws, size_t ws_size,
                              hipStream_t stream) {
  const float* x    = (const float*)d_in[0];
  // d_in[1] = causal_mask (bool) — causality is computed analytically, unused
  const float* Wqkv = (const float*)d_in[2];
  const float* Wout = (const float*)d_in[3];
  const float* bout = (const float*)d_in[4];
  float* out = (float*)d_out;

  char* ws = (char*)d_ws;
  const size_t nqk = (size_t)BB * HH * TT * HD;  // 8M elements per tensor
  __bf16* Qb = (__bf16*)ws; ws += nqk * 2;
  __bf16* Kb = (__bf16*)ws; ws += nqk * 2;
  __bf16* VT = (__bf16*)ws; ws += nqk * 2;
  __bf16* AO = (__bf16*)ws;                      // [B*T, C] bf16

  qkv_kernel<<<dim3(MTOT / 64, (3 * CC) / 64), 128, 0, stream>>>(x, Wqkv, Qb, Kb, VT);
  attn_kernel<<<dim3(TT / 64, BB * HH), 128, 0, stream>>>(Qb, Kb, VT, AO);
  proj_kernel<<<dim3(MTOT / 64, CC / 64), 128, 0, stream>>>(AO, Wout, bout, out);
}
